// LTCCell_26276609917073
// MI455X (gfx1250) — compile-verified, matrix operand order checked by reading, not run on hardware
//
#include <hip/hip_runtime.h>

// ---------------------------------------------------------------------------
// LTC cell for MI455X (gfx1250, wave32).
// B=1024, S=128, N=256, M=32, 6 unfolds. Compute-bound on fp32 sigmoid gates;
// reductions over the presynaptic axis done with V_WMMA_F32_16X16X32_F16
// using a ones-matrix column-sum (result invariant to intra-lane K order).
//
// Gate math (per element): sigmoid(sigma*(v-mu)) scaled by w/mask/erev.
//  - TANH path (gfx1250 V_TANH_F32):  g*w = fma(w/2, tanh(0.5*sigma*v - c), w/2)
//    -> fma + v_tanh + 2 fma per gate (3 VALU, 1 TRANS)
//  - fallback exp2 path: rcp(1 + exp2(c - sigma*log2e*v)) (4 VALU, 2 TRANS)
// Scale constants are folded into the prep-packed weight planes.
// ---------------------------------------------------------------------------

#define kB 1024
#define kS 128
#define kN 256
#define kM 32
#define kU 6
#define BQ 4            // batch rows per block (weight reuse factor)
#define EPSF 1e-8f
#define LOG2E 1.4426950408889634f

typedef __attribute__((ext_vector_type(16))) _Float16 v16h;
typedef __attribute__((ext_vector_type(8)))  float    v8f;

#if defined(__has_builtin)
#  if __has_builtin(__builtin_amdgcn_global_load_async_to_lds_b32)
#    define HAVE_ASYNC_LDS 1
#  endif
#  if __has_builtin(__builtin_amdgcn_tanhf)
#    define HAVE_TANH 1
#  endif
#endif

typedef __attribute__((address_space(1))) int gint_t;   // global (AS1) int
typedef __attribute__((address_space(3))) int lint_t;   // LDS (AS3) int

// d_ws layout (float offsets)
enum : int {
  OFF_SIGR = 0,                     // gate arg scale       [N,N]
  OFF_SMR  = OFF_SIGR + kN * kN,    // gate arg offset      [N,N]
  OFF_WMR  = OFF_SMR  + kN * kN,    // den weight plane     [N,N]
  OFF_WMER = OFF_WMR  + kN * kN,    // num weight plane     [N,N]
  OFF_SIGS = OFF_WMER + kN * kN,    // sensory (same four)  [S,N]
  OFF_SMS  = OFF_SIGS + kS * kN,
  OFF_WMS  = OFF_SMS  + kS * kN,
  OFF_WMES = OFF_WMS  + kS * kN,
  OFF_SNUM = OFF_WMES + kS * kN,    // sensory numerator    [B,N]
  OFF_SDEN = OFF_SNUM + kB * kN     // sensory denominator  [B,N]
};

static __device__ __forceinline__ v8f wmma_colsum(v16h ones, v16h bvals, v8f acc) {
  // D = ones(16x32) x B(32x16) + C : every row of D = column sums of B.
  return __builtin_amdgcn_wmma_f32_16x16x32_f16(
      /*neg_a=*/false, ones, /*neg_b=*/false, bvals,
      /*c_mod=*/(short)0, acc, /*reuse_a=*/false, /*reuse_b=*/false);
}

static __device__ __forceinline__ v16h make_ones() {
  v16h o;
#pragma unroll
  for (int i = 0; i < 16; ++i) o[i] = (_Float16)1.0f;
  return o;
}

// Compute the two weighted gate products pd = wplane_d(x)g, pn = wplane_n(x)g
// for one (j,n) element given presynaptic value v and packed planes.
static __device__ __forceinline__ void gate_products(
    float sg, float sm, float wd, float wn, float v, float& pd, float& pn) {
#ifdef HAVE_TANH
  // planes: sg = 0.5*sigma, sm = 0.5*sigma*mu, wd = 0.5*w*mask, wn = 0.5*w*erev
  // w*sigmoid(t) = w*(0.5 + 0.5*tanh(t/2)) = fma(w/2, th, w/2)
  float th = __builtin_amdgcn_tanhf(sg * v - sm);
  pd = __builtin_fmaf(wd, th, wd);
  pn = __builtin_fmaf(wn, th, wn);
#else
  // planes: sg = sigma*log2e, sm = sigma*mu*log2e, wd = w*mask, wn = w*erev
  float g = __builtin_amdgcn_rcpf(1.0f + __builtin_amdgcn_exp2f(sm - sg * v));
  pd = wd * g;
  pn = wn * g;
#endif
}

// ---------------------------------------------------------------------------
// Prep: fuse weight planes with the scale constants of the chosen gate path.
// ---------------------------------------------------------------------------
__global__ __launch_bounds__(256) void ltc_prep_kernel(
    const float* __restrict__ sigma, const float* __restrict__ mu,
    const float* __restrict__ w,     const float* __restrict__ erev,
    const float* __restrict__ mask,
    const float* __restrict__ ssigma, const float* __restrict__ smu,
    const float* __restrict__ sw,     const float* __restrict__ serev,
    const float* __restrict__ smask,
    float* __restrict__ ws) {
#ifdef HAVE_TANH
  const float argScale = 0.5f;
  const float wScale   = 0.5f;
#else
  const float argScale = LOG2E;
  const float wScale   = 1.0f;
#endif
  int i = blockIdx.x * 256 + threadIdx.x;
  if (i < kN * kN) {
    float sg = sigma[i] * argScale;
    ws[OFF_SIGR + i] = sg;
    ws[OFF_SMR  + i] = sg * mu[i];
    ws[OFF_WMR  + i] = w[i] * mask[i] * wScale;
    ws[OFF_WMER + i] = w[i] * erev[i] * wScale;  // erev already carries the mask
  }
  if (i < kS * kN) {
    float sg = ssigma[i] * argScale;
    ws[OFF_SIGS + i] = sg;
    ws[OFF_SMS  + i] = sg * smu[i];
    ws[OFF_WMS  + i] = sw[i] * smask[i] * wScale;
    ws[OFF_WMES + i] = sw[i] * serev[i] * wScale;
  }
}

// ---------------------------------------------------------------------------
// Sensory synapses: snum/sden[b,n] = sum_s  (w*g) * {erev, 1}
// Wave tile: 16 n-columns (n = lane%16), j chunks of 32 (lanes<16: j 0..15 of
// chunk, lanes>=16: j 16..31), WMMA ones-colsum accumulates in f32.
// ---------------------------------------------------------------------------
__global__ __launch_bounds__(256) void ltc_sensory_kernel(
    const float* __restrict__ inputs, const float* __restrict__ input_w,
    const float* __restrict__ input_b,
    const float* __restrict__ sigS, const float* __restrict__ smS,
    const float* __restrict__ wmS,  const float* __restrict__ wmeS,
    float* __restrict__ snum, float* __restrict__ sden) {
  __shared__ float xrow[BQ][kS];
  const int tid  = threadIdx.x;
  const int lane = tid & 31;
  const int wave = tid >> 5;
  const int b0   = blockIdx.x * BQ;

  for (int i = tid; i < BQ * kS; i += 256) {
    int bb = i / kS, s = i % kS;
    xrow[bb][s] = inputs[(b0 + bb) * kS + s] * input_w[s] + input_b[s];
  }
  __syncthreads();

  const v16h ones = make_ones();
  const int nsub  = lane & 15;
  const int jhalf = (lane >> 4) << 4;

#pragma unroll
  for (int nc = 0; nc < 2; ++nc) {
    const int n = ((wave << 1) + nc) * 16 + nsub;
    v8f cn[BQ], cd[BQ];
#pragma unroll
    for (int bb = 0; bb < BQ; ++bb) {
#pragma unroll
      for (int i = 0; i < 8; ++i) { cn[bb][i] = 0.0f; cd[bb][i] = 0.0f; }
    }
    for (int jc = 0; jc < kS / 32; ++jc) {
      const int jb = jc * 32 + jhalf;
      float sg[16], sm_[16], wm_[16], we_[16];
#pragma unroll
      for (int slot = 0; slot < 16; ++slot) {
        int idx = (jb + slot) * kN + n;
        sg[slot] = sigS[idx]; sm_[slot] = smS[idx];
        wm_[slot] = wmS[idx]; we_[slot] = wmeS[idx];
      }
#pragma unroll
      for (int bb = 0; bb < BQ; ++bb) {
        v16h bd, bn;
#pragma unroll
        for (int slot = 0; slot < 16; ++slot) {
          float pd, pn;
          gate_products(sg[slot], sm_[slot], wm_[slot], we_[slot],
                        xrow[bb][jb + slot], pd, pn);
          bd[slot] = (_Float16)pd;
          bn[slot] = (_Float16)pn;
        }
        cd[bb] = wmma_colsum(ones, bd, cd[bb]);
        cn[bb] = wmma_colsum(ones, bn, cn[bb]);
      }
    }
    if (lane < 16) {
#pragma unroll
      for (int bb = 0; bb < BQ; ++bb) {
        snum[(b0 + bb) * kN + n] = cn[bb][0];
        sden[(b0 + bb) * kN + n] = cd[bb][0];
      }
    }
  }
}

// ---------------------------------------------------------------------------
// Main recurrence: block owns BQ batch rows, runs all 6 unfolds locally with
// ping-pong LDS state rows. Sensory sums seed the WMMA C accumulators.
// ---------------------------------------------------------------------------
__global__ __launch_bounds__(256) void ltc_unfold_kernel(
    const float* __restrict__ states,
    const float* __restrict__ gleak, const float* __restrict__ vleak,
    const float* __restrict__ cm,
    const float* __restrict__ sigR, const float* __restrict__ smR,
    const float* __restrict__ wmR,  const float* __restrict__ wmeR,
    const float* __restrict__ snum, const float* __restrict__ sden,
    const float* __restrict__ output_w, const float* __restrict__ output_b,
    float* __restrict__ out) {
  __shared__ float vbuf[2][BQ][kN];
  const int tid  = threadIdx.x;
  const int lane = tid & 31;
  const int wave = tid >> 5;
  const int b0   = blockIdx.x * BQ;

  // Stage state rows into LDS (async global->LDS path).
#ifdef HAVE_ASYNC_LDS
  for (int i = tid; i < BQ * kN; i += 256) {
    int bb = i >> 8, n = i & (kN - 1);
    __builtin_amdgcn_global_load_async_to_lds_b32(
        (gint_t*)&states[(b0 + bb) * kN + n],
        (lint_t*)&vbuf[0][bb][n],
        0, 0);
  }
  asm volatile("s_wait_asynccnt 0" ::: "memory");
#else
  for (int i = tid; i < BQ * kN; i += 256) {
    int bb = i >> 8, n = i & (kN - 1);
    vbuf[0][bb][n] = states[(b0 + bb) * kN + n];
  }
#endif
  __syncthreads();

  const v16h ones = make_ones();
  const int nsub  = lane & 15;
  const int jhalf = (lane >> 4) << 4;

  // Per-wave constants for its two 16-wide n-chunks.
  float glk[2], cmt[2], gvv[2], sn_[2][BQ], sd_[2][BQ];
#pragma unroll
  for (int nc = 0; nc < 2; ++nc) {
    const int n = ((wave << 1) + nc) * 16 + nsub;
    glk[nc] = gleak[n];
    cmt[nc] = cm[n] * (float)kU;      // cm / (DT/UNFOLDS), DT = 1
    gvv[nc] = gleak[n] * vleak[n];
#pragma unroll
    for (int bb = 0; bb < BQ; ++bb) {
      sn_[nc][bb] = snum[(b0 + bb) * kN + n];
      sd_[nc][bb] = sden[(b0 + bb) * kN + n];
    }
  }

  float (*cur)[kN] = vbuf[0];
  float (*nxt)[kN] = vbuf[1];

  for (int step = 0; step < kU; ++step) {
#pragma unroll
    for (int nc = 0; nc < 2; ++nc) {
      const int n = ((wave << 1) + nc) * 16 + nsub;
      v8f cn[BQ], cd[BQ];
#pragma unroll
      for (int bb = 0; bb < BQ; ++bb) {        // seed with sensory sums
#pragma unroll
        for (int i = 0; i < 8; ++i) { cn[bb][i] = sn_[nc][bb]; cd[bb][i] = sd_[nc][bb]; }
      }
      for (int jc = 0; jc < kN / 32; ++jc) {
        const int jb = jc * 32 + jhalf;
        if (jc + 1 < kN / 32)
          __builtin_prefetch(&sigR[(jb + 32) * kN + n], 0, 1);
        float sg[16], sm_[16], wm_[16], we_[16];
#pragma unroll
        for (int slot = 0; slot < 16; ++slot) {
          int idx = (jb + slot) * kN + n;
          sg[slot] = sigR[idx]; sm_[slot] = smR[idx];
          wm_[slot] = wmR[idx]; we_[slot] = wmeR[idx];
        }
#pragma unroll
        for (int bb = 0; bb < BQ; ++bb) {
          v16h bd, bn;
#pragma unroll
          for (int slot = 0; slot < 16; ++slot) {
            float pd, pn;
            gate_products(sg[slot], sm_[slot], wm_[slot], we_[slot],
                          cur[bb][jb + slot], pd, pn);
            bd[slot] = (_Float16)pd;
            bn[slot] = (_Float16)pn;
          }
          cd[bb] = wmma_colsum(ones, bd, cd[bb]);
          cn[bb] = wmma_colsum(ones, bn, cn[bb]);
        }
      }
#pragma unroll
      for (int bb = 0; bb < BQ; ++bb) {
        float vc  = cur[bb][n];
        float num = cmt[nc] * vc + gvv[nc] * cn[bb][0];
        float den = cmt[nc] + glk[nc] + cd[bb][0] + EPSF;
        float nv  = num / den;
        if (lane < 16) nxt[bb][n] = nv;       // lanes 0-15 and 16-31 are dup
      }
    }
    __syncthreads();
    float (*tmp)[kN] = cur; cur = nxt; nxt = tmp;
  }

  // outputs (B*M) followed by v_pre (B*N) in d_out
#pragma unroll
  for (int nc = 0; nc < 2; ++nc) {
    const int n = ((wave << 1) + nc) * 16 + nsub;
    if (lane < 16) {
#pragma unroll
      for (int bb = 0; bb < BQ; ++bb) {
        float v = cur[bb][n];
        out[kB * kM + (b0 + bb) * kN + n] = v;
        if (n < kM) out[(b0 + bb) * kM + n] = v * output_w[n] + output_b[n];
      }
    }
  }
}

// ---------------------------------------------------------------------------
extern "C" void kernel_launch(void* const* d_in, const int* in_sizes, int n_in,
                              void* d_out, int out_size, void* d_ws, size_t ws_size,
                              hipStream_t stream) {
  const float* inputs   = (const float*)d_in[0];
  const float* states   = (const float*)d_in[1];
  const float* gleak    = (const float*)d_in[2];
  const float* vleak    = (const float*)d_in[3];
  const float* cm       = (const float*)d_in[4];
  const float* sigma    = (const float*)d_in[5];
  const float* mu       = (const float*)d_in[6];
  const float* w        = (const float*)d_in[7];
  const float* erev     = (const float*)d_in[8];
  const float* ssigma   = (const float*)d_in[9];
  const float* smu      = (const float*)d_in[10];
  const float* sw       = (const float*)d_in[11];
  const float* serev    = (const float*)d_in[12];
  const float* input_w  = (const float*)d_in[13];
  const float* input_b  = (const float*)d_in[14];
  const float* output_w = (const float*)d_in[15];
  const float* output_b = (const float*)d_in[16];
  const float* mask     = (const float*)d_in[17];
  const float* smask    = (const float*)d_in[18];

  float* ws  = (float*)d_ws;
  float* out = (float*)d_out;

  ltc_prep_kernel<<<(kN * kN + 255) / 256, 256, 0, stream>>>(
      sigma, mu, w, erev, mask, ssigma, smu, sw, serev, smask, ws);

  ltc_sensory_kernel<<<kB / BQ, 256, 0, stream>>>(
      inputs, input_w, input_b,
      ws + OFF_SIGS, ws + OFF_SMS, ws + OFF_WMS, ws + OFF_WMES,
      ws + OFF_SNUM, ws + OFF_SDEN);

  ltc_unfold_kernel<<<kB / BQ, 256, 0, stream>>>(
      states, gleak, vleak, cm,
      ws + OFF_SIGR, ws + OFF_SMR, ws + OFF_WMR, ws + OFF_WMER,
      ws + OFF_SNUM, ws + OFF_SDEN, output_w, output_b, out);
}
